// SwinTransformer_60344290509071
// MI455X (gfx1250) — matrix operand order repaired and checked by reading, not compiled, not run on hardware
//
#include <hip/hip_runtime.h>
#include <hip/hip_bf16.h>

typedef _Float16 half_t;
typedef __attribute__((ext_vector_type(16))) _Float16 v16h;
typedef __attribute__((ext_vector_type(8)))  _Float16 v8h;
typedef __attribute__((ext_vector_type(8)))  float    v8f;
typedef __attribute__((ext_vector_type(4)))  unsigned int u32x4;
typedef __attribute__((ext_vector_type(8)))  int i32x8;
typedef __attribute__((ext_vector_type(4)))  int i32x4;

#define B_      16
#define DIM_    256
#define NH_     8
#define DH_     32
#define NTOK    64          // tokens per 8x8 window
#define SHIFT_  4
#define NWIN    1024        // B * (64/8)^2
#define HID_    1024
#define NTOKENS 65536       // B*64*64

// ---------------------------------------------------------------------------
// WMMA fragment loaders (CDNA5 v_wmma_f32_16x16x32_f16 layouts, ISA 7.12.2)
// ---------------------------------------------------------------------------
__device__ __forceinline__ v16h load_frag_a(const half_t* base, int ld, int row0, int k0) {
  const int lane = threadIdx.x & 31;
  const half_t* p = base + (size_t)(row0 + (lane & 15)) * ld + k0 + ((lane & 16) ? 8 : 0);
  v8h lo = *reinterpret_cast<const v8h*>(p);
  v8h hi = *reinterpret_cast<const v8h*>(p + 16);
  v16h f;
#pragma unroll
  for (int i = 0; i < 8; ++i) { f[i] = lo[i]; f[i + 8] = hi[i]; }
  return f;
}

__device__ __forceinline__ v16h load_frag_b(const half_t* baseT, int ld, int col0, int k0) {
  const int lane = threadIdx.x & 31;
  const half_t* p = baseT + (size_t)(col0 + (lane & 15)) * ld + k0 + ((lane & 16) ? 8 : 0);
  v8h lo = *reinterpret_cast<const v8h*>(p);
  v8h hi = *reinterpret_cast<const v8h*>(p + 16);
  v16h f;
#pragma unroll
  for (int i = 0; i < 8; ++i) { f[i] = lo[i]; f[i + 8] = hi[i]; }
  return f;
}

__device__ __forceinline__ v8f wmma16(v16h a, v16h b, v8f c) {
  return __builtin_amdgcn_wmma_f32_16x16x32_f16(false, a, false, b, (short)0, c, false, false);
}

// ---------------------------------------------------------------------------
// The TDM builtin carries the LDS destination as opaque descriptor bits, so the
// compiler cannot see that the __shared__ buffer is written. Escape the pointer
// into an opaque asm with a memory clobber so loads from it are not folded away.
// ---------------------------------------------------------------------------
__device__ __forceinline__ void lds_publish(const void* p) {
  asm volatile("" : : "v"(p) : "memory");
}

// ---------------------------------------------------------------------------
// Tensor Data Mover: DMA a 2D tile (rows x cols f16, row-major, stride in elems)
// from global memory into LDS at lds_off. Issued by one wave; caller must
// s_wait_tensorcnt + barrier + lds_publish before consuming. D# per ISA ch.8.
// ---------------------------------------------------------------------------
__device__ __forceinline__ void tdm_load_2d_f16(unsigned lds_off, const half_t* gsrc,
                                                unsigned rows, unsigned cols,
                                                unsigned stride_elems) {
  const unsigned long long ga = (unsigned long long)(size_t)gsrc;
  u32x4 g0;
  g0[0] = 1u;                                            // count=1, user descriptor
  g0[1] = lds_off;                                       // lds_addr (bytes)
  g0[2] = (unsigned)(ga & 0xffffffffull);                // global_addr[31:0]
  g0[3] = (unsigned)((ga >> 32) & 0x1ffffffull) | (2u << 30); // addr[56:32] | type=2
  i32x8 g1;
  g1[0] = (int)(1u << 16);                               // wg_mask=0, data_size=1 (2B)
  g1[1] = (int)((cols & 0xffffu) << 16);                 // tensor_dim0[15:0]
  g1[2] = (int)(((cols >> 16) & 0xffffu) | ((rows & 0xffffu) << 16)); // dim0 hi | dim1 lo
  g1[3] = (int)(((rows >> 16) & 0xffffu) | ((cols & 0xffffu) << 16)); // dim1 hi | tile_dim0
  g1[4] = (int)(rows & 0xffffu);                         // tile_dim1 | tile_dim2=0
  g1[5] = (int)stride_elems;                             // tensor_dim0_stride[31:0]
  g1[6] = 0;                                             // stride hi | dim1_stride lo
  g1[7] = 0;
  i32x4 z4 = {0, 0, 0, 0};
#if defined(__clang_major__) && (__clang_major__ >= 23)
  i32x8 z8 = {0, 0, 0, 0, 0, 0, 0, 0};
  __builtin_amdgcn_tensor_load_to_lds(g0, g1, z4, z4, z8, 0);
#else
  __builtin_amdgcn_tensor_load_to_lds(g0, g1, z4, z4, 0);
#endif
}

// ---------------------------------------------------------------------------
// Weight fp32 [R,C] -> f16 transposed [C,R]
// ---------------------------------------------------------------------------
__global__ void w_to_f16T_kernel(const float* __restrict__ src, half_t* __restrict__ dst,
                                 int R, int C) {
  int idx = blockIdx.x * 256 + threadIdx.x;
  if (idx >= R * C) return;
  int r = idx / C, c = idx % C;
  dst[(size_t)c * R + r] = (half_t)src[idx];
}

// ---------------------------------------------------------------------------
// LN1 + (shift) + window partition -> xw f16 [1024 win][64 tok][256]
// ---------------------------------------------------------------------------
__global__ void ln_partition_kernel(const float* __restrict__ x, const float* __restrict__ g,
                                    const float* __restrict__ bt, half_t* __restrict__ xw,
                                    int shifted) {
  const int lane = threadIdx.x & 31;
  const int tok = blockIdx.x * 8 + (threadIdx.x >> 5);
  const int b = tok >> 12, hw = tok & 4095, h = hw >> 6, w = hw & 63;
  const float* xp = x + (size_t)tok * DIM_ + lane * 8;
  float v[8]; float s = 0.f, ss = 0.f;
#pragma unroll
  for (int i = 0; i < 8; ++i) { v[i] = xp[i]; s += v[i]; ss += v[i] * v[i]; }
#pragma unroll
  for (int m = 16; m >= 1; m >>= 1) { s += __shfl_xor(s, m, 32); ss += __shfl_xor(ss, m, 32); }
  const float mean = s * (1.f / DIM_);
  const float rstd = rsqrtf(ss * (1.f / DIM_) - mean * mean + 1e-5f);
  int dh = h, dw = w;
  if (shifted) { dh = (h - SHIFT_) & 63; dw = (w - SHIFT_) & 63; }
  const int gwin = b * 64 + (dh >> 3) * 8 + (dw >> 3);
  const int loc = (dh & 7) * 8 + (dw & 7);
  half_t* out = xw + ((size_t)gwin * NTOK + loc) * DIM_ + lane * 8;
#pragma unroll
  for (int i = 0; i < 8; ++i) {
    int c = lane * 8 + i;
    out[i] = (half_t)((v[i] - mean) * rstd * g[c] + bt[c]);
  }
}

// ---------------------------------------------------------------------------
// QKV GEMM: TDM-stage window A [64,256] into LDS, weights stream from L2
// scatter q,k (row-major) / v (transposed)
// ---------------------------------------------------------------------------
__global__ void qkv_gemm_kernel(const half_t* __restrict__ xw, const half_t* __restrict__ wT,
                                const float* __restrict__ bias, half_t* __restrict__ q,
                                half_t* __restrict__ k, half_t* __restrict__ vt) {
  __shared__ half_t As[NTOK * DIM_];        // 32 KB, LDS offset 0
  const int win = blockIdx.x;
  const int wid = threadIdx.x >> 5, lane = threadIdx.x & 31;
  if (threadIdx.x < 32) {
    tdm_load_2d_f16(0, xw + (size_t)win * NTOK * DIM_, NTOK, DIM_, DIM_);
    __builtin_amdgcn_s_wait_tensorcnt(0);
  }
  __syncthreads();
  lds_publish(As);
  for (int t = wid; t < 4 * 48; t += 8) {
    const int mt = t / 48, nt = t % 48;
    v8f acc = {};
#pragma unroll
    for (int k0 = 0; k0 < DIM_; k0 += 32)
      acc = wmma16(load_frag_a(As, DIM_, mt * 16, k0), load_frag_b(wT, DIM_, nt * 16, k0), acc);
    const int col = nt * 16 + (lane & 15);
    const int which = col >> 8, rem = col & 255, head = rem >> 5, d = rem & 31;
    const float bv = bias[col];
    const size_t hb = (size_t)win * NH_ + head;
#pragma unroll
    for (int i = 0; i < 8; ++i) {
      const int m = mt * 16 + i + ((lane & 16) ? 8 : 0);
      const half_t val = (half_t)(acc[i] + bv);
      if (which == 0)      q[(hb * NTOK + m) * DH_ + d] = val;
      else if (which == 1) k[(hb * NTOK + m) * DH_ + d] = val;
      else                 vt[(hb * DH_ + d) * NTOK + m] = val;
    }
  }
}

// ---------------------------------------------------------------------------
// Attention per (window, head): dots (WMMA) + bias + mask -> LDS softmax -> P@V
// ---------------------------------------------------------------------------
__global__ void attn_kernel(const half_t* __restrict__ q, const half_t* __restrict__ k,
                            const half_t* __restrict__ vt, const float* __restrict__ rpb,
                            half_t* __restrict__ aout, int shifted) {
  __shared__ float  dotsS[NTOK * NTOK];
  __shared__ half_t attnS[NTOK * NTOK];
  const int whid = blockIdx.x;
  const int win = whid >> 3, head = whid & 7;
  const int wid = threadIdx.x >> 5, lane = threadIdx.x & 31;
  const half_t* qp = q + (size_t)whid * NTOK * DH_;
  const half_t* kp = k + (size_t)whid * NTOK * DH_;
  const float scale = 0.17677669529663687f;   // 32^-0.5
  const int winb = win & 63;
  const int wh0 = (winb >> 3) * 8, ww0 = (winb & 7) * 8;

  v16h afrag = load_frag_a(qp, DH_, wid * 16, 0);
#pragma unroll
  for (int nt = 0; nt < 4; ++nt) {
    v8f acc = {};
    acc = wmma16(afrag, load_frag_b(kp, DH_, nt * 16, 0), acc);
    const int j = nt * 16 + (lane & 15);
    const int rj = j >> 3, cj = j & 7;
    int regj = 0;
    if (shifted)
      regj = ((wh0 + rj) >= 56 ? ((wh0 + rj) >= 60 ? 2 : 1) : 0) * 3 +
             ((ww0 + cj) >= 56 ? ((ww0 + cj) >= 60 ? 2 : 1) : 0);
#pragma unroll
    for (int i = 0; i < 8; ++i) {
      const int m = wid * 16 + i + ((lane & 16) ? 8 : 0);
      const int ri = m >> 3, ci = m & 7;
      const int idx = (ri - rj + 7) * 15 + (ci - cj + 7);
      float val = acc[i] * scale + rpb[idx * NH_ + head];
      if (shifted) {
        int regi = ((wh0 + ri) >= 56 ? ((wh0 + ri) >= 60 ? 2 : 1) : 0) * 3 +
                   ((ww0 + ci) >= 56 ? ((ww0 + ci) >= 60 ? 2 : 1) : 0);
        if (regi != regj) val -= 100.0f;
      }
      dotsS[m * NTOK + j] = val;
    }
  }
  __syncthreads();

  if (threadIdx.x < NTOK) {
    const int row = threadIdx.x;
    float mx = -1e30f;
    for (int j = 0; j < NTOK; ++j) mx = fmaxf(mx, dotsS[row * NTOK + j]);
    float sum = 0.f;
    for (int j = 0; j < NTOK; ++j) sum += __expf(dotsS[row * NTOK + j] - mx);
    const float inv = 1.f / sum;
    for (int j = 0; j < NTOK; ++j)
      attnS[row * NTOK + j] = (half_t)(__expf(dotsS[row * NTOK + j] - mx) * inv);
  }
  __syncthreads();

  const half_t* vp = vt + (size_t)whid * DH_ * NTOK;   // [32 dh][64 tok]
#pragma unroll
  for (int nt = 0; nt < 2; ++nt) {
    v8f acc = {};
#pragma unroll
    for (int k0 = 0; k0 < NTOK; k0 += 32)
      acc = wmma16(load_frag_a(attnS, NTOK, wid * 16, k0),
                   load_frag_b(vp, NTOK, nt * 16, k0), acc);
    const int d = nt * 16 + (lane & 15);
#pragma unroll
    for (int i = 0; i < 8; ++i) {
      const int m = wid * 16 + i + ((lane & 16) ? 8 : 0);
      aout[((size_t)win * NTOK + m) * DIM_ + head * DH_ + d] = (half_t)acc[i];
    }
  }
}

// ---------------------------------------------------------------------------
// Out-proj + window-reverse + unshift + residual into xcur (fp32); TDM-staged A
// ---------------------------------------------------------------------------
__global__ void proj_residual_kernel(const half_t* __restrict__ aout,
                                     const half_t* __restrict__ wT,
                                     const float* __restrict__ bias,
                                     float* __restrict__ xcur, int shifted) {
  __shared__ half_t As[NTOK * DIM_];        // 32 KB, LDS offset 0
  const int win = blockIdx.x;
  const int wid = threadIdx.x >> 5, lane = threadIdx.x & 31;
  const int b = win >> 6, winb = win & 63;
  const int wh0 = (winb >> 3) * 8, ww0 = (winb & 7) * 8;
  if (threadIdx.x < 32) {
    tdm_load_2d_f16(0, aout + (size_t)win * NTOK * DIM_, NTOK, DIM_, DIM_);
    __builtin_amdgcn_s_wait_tensorcnt(0);
  }
  __syncthreads();
  lds_publish(As);
  for (int t = wid; t < 4 * 16; t += 8) {
    const int mt = t >> 4, nt = t & 15;
    v8f acc = {};
#pragma unroll
    for (int k0 = 0; k0 < DIM_; k0 += 32)
      acc = wmma16(load_frag_a(As, DIM_, mt * 16, k0), load_frag_b(wT, DIM_, nt * 16, k0), acc);
    const int col = nt * 16 + (lane & 15);
    const float bv = bias[col];
#pragma unroll
    for (int i = 0; i < 8; ++i) {
      const int m = mt * 16 + i + ((lane & 16) ? 8 : 0);
      int h = wh0 + (m >> 3), w = ww0 + (m & 7);
      if (shifted) { h = (h + SHIFT_) & 63; w = (w + SHIFT_) & 63; }
      const size_t idx = (((size_t)b * 4096) + h * 64 + w) * DIM_ + col;
      xcur[idx] = xcur[idx] + acc[i] + bv;
    }
  }
}

// ---------------------------------------------------------------------------
// LN(1e-5) then LN(1e-6) -> hin f16 [NTOKENS,256]
// ---------------------------------------------------------------------------
__global__ void double_ln_kernel(const float* __restrict__ x,
                                 const float* __restrict__ g2, const float* __restrict__ b2,
                                 const float* __restrict__ g3, const float* __restrict__ b3,
                                 half_t* __restrict__ hin) {
  const int lane = threadIdx.x & 31;
  const int tok = blockIdx.x * 8 + (threadIdx.x >> 5);
  const float* xp = x + (size_t)tok * DIM_ + lane * 8;
  float v[8]; float s = 0.f, ss = 0.f;
#pragma unroll
  for (int i = 0; i < 8; ++i) { v[i] = xp[i]; s += v[i]; ss += v[i] * v[i]; }
#pragma unroll
  for (int m = 16; m >= 1; m >>= 1) { s += __shfl_xor(s, m, 32); ss += __shfl_xor(ss, m, 32); }
  float mean = s * (1.f / DIM_);
  float rstd = rsqrtf(ss * (1.f / DIM_) - mean * mean + 1e-5f);
  s = 0.f; ss = 0.f;
#pragma unroll
  for (int i = 0; i < 8; ++i) {
    int c = lane * 8 + i;
    v[i] = (v[i] - mean) * rstd * g2[c] + b2[c];
    s += v[i]; ss += v[i] * v[i];
  }
#pragma unroll
  for (int m = 16; m >= 1; m >>= 1) { s += __shfl_xor(s, m, 32); ss += __shfl_xor(ss, m, 32); }
  mean = s * (1.f / DIM_);
  rstd = rsqrtf(ss * (1.f / DIM_) - mean * mean + 1e-6f);
  half_t* out = hin + (size_t)tok * DIM_ + lane * 8;
#pragma unroll
  for (int i = 0; i < 8; ++i) {
    int c = lane * 8 + i;
    out[i] = (half_t)((v[i] - mean) * rstd * g3[c] + b3[c]);
  }
}

// ---------------------------------------------------------------------------
// MLP GEMM1 + exact GELU: [64-row chunk,256] x [256,1024]; TDM-staged A
// ---------------------------------------------------------------------------
__global__ void mlp1_kernel(const half_t* __restrict__ hin, const half_t* __restrict__ w1T,
                            const float* __restrict__ b1, half_t* __restrict__ hmid) {
  __shared__ half_t As[64 * DIM_];          // 32 KB, LDS offset 0
  const int chunk = blockIdx.x;
  const int wid = threadIdx.x >> 5, lane = threadIdx.x & 31;
  if (threadIdx.x < 32) {
    tdm_load_2d_f16(0, hin + (size_t)chunk * 64 * DIM_, 64, DIM_, DIM_);
    __builtin_amdgcn_s_wait_tensorcnt(0);
  }
  __syncthreads();
  lds_publish(As);
  for (int t = wid; t < 4 * 64; t += 8) {
    const int mt = t >> 6, nt = t & 63;
    v8f acc = {};
#pragma unroll
    for (int k0 = 0; k0 < DIM_; k0 += 32)
      acc = wmma16(load_frag_a(As, DIM_, mt * 16, k0), load_frag_b(w1T, DIM_, nt * 16, k0), acc);
    const int col = nt * 16 + (lane & 15);
    const float bv = b1[col];
#pragma unroll
    for (int i = 0; i < 8; ++i) {
      const int m = mt * 16 + i + ((lane & 16) ? 8 : 0);
      float z = acc[i] + bv;
      z = 0.5f * z * (1.0f + erff(z * 0.7071067811865475f));
      hmid[((size_t)chunk * 64 + m) * HID_ + col] = (half_t)z;
    }
  }
}

// ---------------------------------------------------------------------------
// MLP GEMM2 + residual: [64-chunk,1024] x [1024,256] + res -> out fp32
// K staged in 4 x (64x256) TDM chunks; persistent per-wave accumulators
// ---------------------------------------------------------------------------
__global__ void mlp2_kernel(const half_t* __restrict__ hmid, const half_t* __restrict__ w2T,
                            const float* __restrict__ b2, const float* __restrict__ res,
                            float* __restrict__ out) {
  __shared__ half_t As[64 * 256];           // 32 KB K-chunk, LDS offset 0
  const int chunk = blockIdx.x;
  const int wid = threadIdx.x >> 5, lane = threadIdx.x & 31;
  const half_t* A = hmid + (size_t)chunk * 64 * HID_;
  v8f acc[8];
#pragma unroll
  for (int j = 0; j < 8; ++j) acc[j] = (v8f){};

  for (int kc = 0; kc < 4; ++kc) {          // K chunks of 256
    __syncthreads();                        // previous chunk fully consumed
    if (threadIdx.x < 32) {
      tdm_load_2d_f16(0, A + kc * 256, 64, 256, HID_);  // 64 rows, stride 1024
      __builtin_amdgcn_s_wait_tensorcnt(0);
    }
    __syncthreads();
    lds_publish(As);
#pragma unroll
    for (int j = 0; j < 8; ++j) {
      const int t = wid + j * 8;
      const int mt = t >> 4, nt = t & 15;
#pragma unroll
      for (int k0 = 0; k0 < 256; k0 += 32)
        acc[j] = wmma16(load_frag_a(As, 256, mt * 16, k0),
                        load_frag_b(w2T, HID_, nt * 16, kc * 256 + k0), acc[j]);
    }
  }
#pragma unroll
  for (int j = 0; j < 8; ++j) {
    const int t = wid + j * 8;
    const int mt = t >> 4, nt = t & 15;
    const int col = nt * 16 + (lane & 15);
    const float bv = b2[col];
#pragma unroll
    for (int i = 0; i < 8; ++i) {
      const int m = mt * 16 + i + ((lane & 16) ? 8 : 0);
      const size_t row = (size_t)chunk * 64 + m;
      out[row * DIM_ + col] = res[row * DIM_ + col] + acc[j][i] + bv;
    }
  }
}

// ---------------------------------------------------------------------------
extern "C" void kernel_launch(void* const* d_in, const int* in_sizes, int n_in,
                              void* d_out, int out_size, void* d_ws, size_t ws_size,
                              hipStream_t stream) {
  (void)in_sizes; (void)n_in; (void)out_size; (void)ws_size;
  uint8_t* w = (uint8_t*)d_ws;
  size_t off = 0;
  float*  xcur = (float*)(w + off);  off += (size_t)NTOKENS * DIM_ * 4;   // 64 MB
  half_t* xw   = (half_t*)(w + off); off += (size_t)NTOKENS * DIM_ * 2;
  half_t* qf   = (half_t*)(w + off); off += (size_t)NTOKENS * DIM_ * 2;
  half_t* kf   = (half_t*)(w + off); off += (size_t)NTOKENS * DIM_ * 2;
  half_t* vt   = (half_t*)(w + off); off += (size_t)NTOKENS * DIM_ * 2;
  half_t* aout = (half_t*)(w + off); off += (size_t)NTOKENS * DIM_ * 2;
  half_t* hin  = (half_t*)(w + off); off += (size_t)NTOKENS * DIM_ * 2;
  half_t* hmid = (half_t*)(w + off); off += (size_t)NTOKENS * HID_ * 2;   // 128 MB
  half_t* wqkvT = (half_t*)(w + off); off += (size_t)768 * 256 * 2;
  half_t* woutT = (half_t*)(w + off); off += (size_t)256 * 256 * 2;
  half_t* w1T   = (half_t*)(w + off); off += (size_t)1024 * 256 * 2;
  half_t* w2T   = (half_t*)(w + off); off += (size_t)256 * 1024 * 2;

  hipMemcpyAsync(xcur, d_in[0], (size_t)NTOKENS * DIM_ * 4,
                 hipMemcpyDeviceToDevice, stream);

  for (int blk = 0; blk < 2; ++blk) {
    const int p = 1 + blk * 15;
    const float* n1g   = (const float*)d_in[p + 0];
    const float* n1b   = (const float*)d_in[p + 1];
    const float* qkv_w = (const float*)d_in[p + 2];
    const float* qkv_b = (const float*)d_in[p + 3];
    const float* rpb   = (const float*)d_in[p + 4];
    const float* out_w = (const float*)d_in[p + 5];
    const float* out_b = (const float*)d_in[p + 6];
    const float* n2g   = (const float*)d_in[p + 7];
    const float* n2b   = (const float*)d_in[p + 8];
    const float* lng   = (const float*)d_in[p + 9];
    const float* lnb   = (const float*)d_in[p + 10];
    const float* w1    = (const float*)d_in[p + 11];
    const float* b1    = (const float*)d_in[p + 12];
    const float* w2    = (const float*)d_in[p + 13];
    const float* b2    = (const float*)d_in[p + 14];
    const int shifted = blk;

    w_to_f16T_kernel<<<(256 * 768 + 255) / 256, 256, 0, stream>>>(qkv_w, wqkvT, 256, 768);
    w_to_f16T_kernel<<<(256 * 256 + 255) / 256, 256, 0, stream>>>(out_w, woutT, 256, 256);
    w_to_f16T_kernel<<<(256 * 1024 + 255) / 256, 256, 0, stream>>>(w1, w1T, 256, 1024);
    w_to_f16T_kernel<<<(1024 * 256 + 255) / 256, 256, 0, stream>>>(w2, w2T, 1024, 256);

    ln_partition_kernel<<<NTOKENS / 8, 256, 0, stream>>>(xcur, n1g, n1b, xw, shifted);
    qkv_gemm_kernel<<<NWIN, 256, 0, stream>>>(xw, wqkvT, qkv_b, qf, kf, vt);
    attn_kernel<<<NWIN * NH_, 128, 0, stream>>>(qf, kf, vt, rpb, aout, shifted);
    proj_residual_kernel<<<NWIN, 256, 0, stream>>>(aout, woutT, out_b, xcur, shifted);
    double_ln_kernel<<<NTOKENS / 8, 256, 0, stream>>>(xcur, n2g, n2b, lng, lnb, hin);
    mlp1_kernel<<<NTOKENS / 64, 256, 0, stream>>>(hin, w1T, b1, hmid);
    float* tgt = (blk == 1) ? (float*)d_out : xcur;
    mlp2_kernel<<<NTOKENS / 64, 256, 0, stream>>>(hmid, w2T, b2, xcur, tgt);
  }
}
